// FeatureAttention_79886391706031
// MI455X (gfx1250) — matrix operand order repaired and checked by reading, not compiled, not run on hardware
//
#include <hip/hip_runtime.h>
#include <cstdint>

// ---------------------------------------------------------------------------
// FeatureAttention for MI455X (gfx1250, wave32, WMMA).
//   B=4, C=256, CQK=32, N=64*64=4096
//   d_out = [ out (B*C*N f32) | attention (B*N*N f32) ]
// GEMMs use v_wmma_f32_16x16x32_bf16 with operands staged in layouts that map
// directly onto the ISA's WMMA register fragments (no LDS staging needed).
// The 268 MB attention matrix is the HBM-bound term: it is written once
// (energy), softmaxed in place, and read with 4x tile reuse (out GEMM).
// ---------------------------------------------------------------------------

typedef __attribute__((ext_vector_type(16))) __bf16        v16bf;
typedef __attribute__((ext_vector_type(8)))  float         v8f;
typedef __attribute__((ext_vector_type(8)))  unsigned short v8u16;
typedef __attribute__((ext_vector_type(4)))  float         v4f;

constexpr int B_   = 4;
constexpr int C_   = 256;
constexpr int CQK_ = 32;
constexpr int N_   = 4096;

static __device__ __forceinline__ unsigned short f32_to_bf16(float f) {
    unsigned int u = __float_as_uint(f);
    u += 0x7FFFu + ((u >> 16) & 1u);          // round-to-nearest-even
    return (unsigned short)(u >> 16);
}

union Frag16 {                                 // 16 bf16 halves = one WMMA A/B fragment
    v16bf v;
    v8u16 h[2];
    unsigned short u[16];
};

// ---------------------------------------------------------------------------
// Projections (tiny vs. the N^2 work): f32 dot products, bf16 outputs.
// pq , pkT : [B][N][32] bf16   (rows contiguous in d -> direct WMMA fragments)
// pv       : [B][C][N]  bf16   (rows contiguous in n -> direct WMMA A frags)
// ---------------------------------------------------------------------------
__global__ void fa_proj_qk_kernel(const float* __restrict__ q,
                                  const float* __restrict__ k,
                                  const float* __restrict__ Wq, const float* __restrict__ bq,
                                  const float* __restrict__ Wk, const float* __restrict__ bk,
                                  unsigned short* __restrict__ pq,
                                  unsigned short* __restrict__ pkT) {
    const int b    = blockIdx.z;
    const int y    = blockIdx.y;               // 0..63 : [0,32) -> Q, [32,64) -> K
    const bool isq = (y < CQK_);
    const int d    = isq ? y : (y - CQK_);
    const int n    = blockIdx.x * blockDim.x + threadIdx.x;

    const float* src  = (isq ? q : k) + (size_t)b * C_ * N_ + n;   // coalesced over n
    const float* Wrow = (isq ? Wq : Wk) + (size_t)d * C_;          // uniform -> scalar loads
    float acc = isq ? bq[d] : bk[d];
#pragma unroll 8
    for (int c = 0; c < C_; ++c)
        acc = fmaf(Wrow[c], src[(size_t)c * N_], acc);

    unsigned short* dst = (isq ? pq : pkT) + ((size_t)b * N_ + n) * CQK_ + d;
    *dst = f32_to_bf16(acc);
}

__global__ void fa_proj_v_kernel(const float* __restrict__ v,
                                 const float* __restrict__ Wv, const float* __restrict__ bv,
                                 unsigned short* __restrict__ pv) {
    const int b = blockIdx.z;
    const int c = blockIdx.y;
    const int n = blockIdx.x * blockDim.x + threadIdx.x;

    const float* src  = v + (size_t)b * C_ * N_ + n;
    const float* Wrow = Wv + (size_t)c * C_;
    float acc = bv[c];
#pragma unroll 8
    for (int cc = 0; cc < C_; ++cc)
        acc = fmaf(Wrow[cc], src[(size_t)cc * N_], acc);

    pv[((size_t)b * C_ + c) * N_ + n] = f32_to_bf16(acc);          // coalesced
}

// ---------------------------------------------------------------------------
// Energy: E[b][n][m] = sum_d pq[b][n][d] * pkT[b][m][d]
// K = CQK = 32 -> exactly ONE v_wmma_f32_16x16x32_bf16 per 16x16 tile.
// Wave computes a 16(n) x 64(m) strip (A fragment reused across 4 WMMAs).
// Fragment mapping used here (wave32; A/B k order chosen by this code):
//   A lane L: row = L&15,  half  h   -> K=(L>>4)*16+h   (contiguous!)
//   B lane L: col = L&15,  half  h   -> K=(L>>4)*16+h   (contiguous!)
//   D lane L, vgpr v: row = v + 8*(L>>4), col = L&15
// ---------------------------------------------------------------------------
__global__ void fa_energy_kernel(const unsigned short* __restrict__ pq,
                                 const unsigned short* __restrict__ pkT,
                                 float* __restrict__ att) {
    const int lane   = threadIdx.x & 31;
    const int wave   = threadIdx.x >> 5;
    const int b      = blockIdx.z;
    const int n0     = (blockIdx.y * 8 + wave) * 16;
    const int m_base = blockIdx.x * 64;
    const int l15    = lane & 15;
    const int hi     = lane >> 4;
    const int kbase  = hi * 8;
    const int kstart = hi * 16;

    // A fragment: two 16B chunks from the pq row of this lane.
    Frag16 a;
    const unsigned short* arow = pq + ((size_t)b * N_ + (n0 + l15)) * CQK_;
    a.h[0] = *(const v8u16*)(arow + kstart);
    a.h[1] = *(const v8u16*)(arow + kstart + 8);

#pragma unroll
    for (int t = 0; t < 4; ++t) {
        const int m0 = m_base + t * 16;
        // B fragment: one contiguous 32B chunk of pkT row (32B aligned).
        const unsigned short* brow =
            pkT + ((size_t)b * N_ + (m0 + l15)) * CQK_ + kstart;
        v16bf bm = *(const v16bf*)brow;

        v8f c = {0.f, 0.f, 0.f, 0.f, 0.f, 0.f, 0.f, 0.f};
        c = __builtin_amdgcn_wmma_f32_16x16x32_bf16(
                false, a.v, false, bm, (short)0, c, false, false);

        float* drow = att + ((size_t)b * N_ + (n0 + hi * 8)) * N_ + m0 + l15;
#pragma unroll
        for (int vv = 0; vv < 8; ++vv)
            drow[(size_t)vv * N_] = c[vv];
    }
}

// ---------------------------------------------------------------------------
// Row softmax over the last axis, in place on the attention region.
// One 256-thread block per row (B*N rows), 16 values/thread held in VGPRs.
// ---------------------------------------------------------------------------
__global__ void fa_softmax_kernel(float* __restrict__ att) {
    const int tid = threadIdx.x;
    float* row = att + (size_t)blockIdx.x * N_;

    float vals[16];
    float mx = -3.402823466e38f;
#pragma unroll
    for (int j = 0; j < 16; ++j) {
        vals[j] = row[tid + j * 256];
        mx = fmaxf(mx, vals[j]);
    }

    __shared__ float red[256];
    red[tid] = mx;
    __syncthreads();
    for (int s = 128; s > 0; s >>= 1) {
        if (tid < s) red[tid] = fmaxf(red[tid], red[tid + s]);
        __syncthreads();
    }
    mx = red[0];
    __syncthreads();

    float sum = 0.f;
#pragma unroll
    for (int j = 0; j < 16; ++j) {
        vals[j] = __expf(vals[j] - mx);
        sum += vals[j];
    }
    red[tid] = sum;
    __syncthreads();
    for (int s = 128; s > 0; s >>= 1) {
        if (tid < s) red[tid] += red[tid + s];
        __syncthreads();
    }
    const float inv = 1.0f / red[0];
#pragma unroll
    for (int j = 0; j < 16; ++j)
        row[tid + j * 256] = vals[j] * inv;
}

// ---------------------------------------------------------------------------
// Out[b][c][m] = gamma * sum_n pv[b][c][n] * att[b][m][n] + x[b][c][m]
// Wave tile: 64 rows (c) x 16 cols (m). Per K-step (32):
//   - 4 A fragments from pv (bf16, L2-resident, cheap)
//   - 1 B fragment from att (16 contiguous f32 per lane -> bf16 in regs),
//     SHARED by all 4 WMMAs -> att traffic & cvt VALU cut 4x vs 16-row tiles.
// Epilogue fuses gamma*acc + x; out/x use non-temporal policy so they don't
// evict the L2-resident att slice being re-read.
// ---------------------------------------------------------------------------
__global__ void fa_out_kernel(const unsigned short* __restrict__ pv,
                              const float* __restrict__ att,
                              const float* __restrict__ x,
                              const float* __restrict__ gamma,
                              float* __restrict__ out) {
    const int lane   = threadIdx.x & 31;
    const int wave   = threadIdx.x >> 5;
    const int b      = blockIdx.z;
    const int c0     = blockIdx.y * 64;                 // 4 c-tiles per wave
    const int m0     = (blockIdx.x * 8 + wave) * 16;    // one m-tile per wave
    const int l15    = lane & 15;
    const int hi     = lane >> 4;
    const int kbase  = hi * 8;
    const int kstart = hi * 16;

    v8f acc[4];
#pragma unroll
    for (int t = 0; t < 4; ++t) {
        v8f z = {0.f, 0.f, 0.f, 0.f, 0.f, 0.f, 0.f, 0.f};
        acc[t] = z;
    }

    const unsigned short* arow[4];
#pragma unroll
    for (int t = 0; t < 4; ++t)
        arow[t] = pv + ((size_t)b * C_ + (c0 + t * 16 + l15)) * N_;

    const float* brow = att + ((size_t)b * N_ + (m0 + l15)) * N_ + kstart;

    for (int k0 = 0; k0 < N_; k0 += 32) {
        // One shared B fragment: 16 contiguous f32 of this lane's att row.
        Frag16 bb;
        const float* ar = brow + k0;                    // 64B aligned
#pragma unroll
        for (int i = 0; i < 4; ++i) {
            v4f f = *(const v4f*)(ar + i * 4);
            bb.u[i * 4 + 0] = f32_to_bf16(f[0]);
            bb.u[i * 4 + 1] = f32_to_bf16(f[1]);
            bb.u[i * 4 + 2] = f32_to_bf16(f[2]);
            bb.u[i * 4 + 3] = f32_to_bf16(f[3]);
        }

#pragma unroll
        for (int t = 0; t < 4; ++t) {
            Frag16 a;
            a.h[0] = *(const v8u16*)(arow[t] + k0 + kstart);
            a.h[1] = *(const v8u16*)(arow[t] + k0 + kstart + 8);
            acc[t] = __builtin_amdgcn_wmma_f32_16x16x32_bf16(
                         false, a.v, false, bb.v, (short)0, acc[t], false, false);
        }
    }

    const float g  = gamma[0];
    const int   mm = m0 + l15;
#pragma unroll
    for (int t = 0; t < 4; ++t) {
        const int cc = c0 + t * 16 + hi * 8;            // + vv below
        float*       orow = out + ((size_t)b * C_ + cc) * N_ + mm;
        const float* xrow = x   + ((size_t)b * C_ + cc) * N_ + mm;
#pragma unroll
        for (int vv = 0; vv < 8; ++vv) {
            float xv = __builtin_nontemporal_load(xrow + (size_t)vv * N_);
            __builtin_nontemporal_store(g * acc[t][vv] + xv,
                                        orow + (size_t)vv * N_);
        }
    }
}

// ---------------------------------------------------------------------------
extern "C" void kernel_launch(void* const* d_in, const int* in_sizes, int n_in,
                              void* d_out, int out_size, void* d_ws, size_t ws_size,
                              hipStream_t stream) {
    const float* query = (const float*)d_in[0];
    const float* key   = (const float*)d_in[1];
    const float* value = (const float*)d_in[2];
    const float* x     = (const float*)d_in[3];
    const float* Wq    = (const float*)d_in[4];
    const float* bq    = (const float*)d_in[5];
    const float* Wk    = (const float*)d_in[6];
    const float* bk    = (const float*)d_in[7];
    const float* Wv    = (const float*)d_in[8];
    const float* bv    = (const float*)d_in[9];
    const float* gamma = (const float*)d_in[10];

    float* out = (float*)d_out;                          // B*C*N
    float* att = out + (size_t)B_ * C_ * N_;             // B*N*N

    // Workspace (10 MiB, fully overwritten each call):
    unsigned short* pq  = (unsigned short*)d_ws;                       // [B][N][32]
    unsigned short* pkT = pq  + (size_t)B_ * N_ * CQK_;                // [B][N][32]
    unsigned short* pv  = pkT + (size_t)B_ * N_ * CQK_;                // [B][C][N]

    fa_proj_qk_kernel<<<dim3(N_ / 256, 2 * CQK_, B_), 256, 0, stream>>>(
        query, key, Wq, bq, Wk, bk, pq, pkT);
    fa_proj_v_kernel<<<dim3(N_ / 256, C_, B_), 256, 0, stream>>>(
        value, Wv, bv, pv);
    fa_energy_kernel<<<dim3(N_ / 64, N_ / 128, B_), 256, 0, stream>>>(
        pq, pkT, att);
    fa_softmax_kernel<<<dim3(B_ * N_), 256, 0, stream>>>(att);
    fa_out_kernel<<<dim3(N_ / 128, C_ / 64, B_), 256, 0, stream>>>(
        pv, att, x, gamma, out);

    (void)in_sizes; (void)n_in; (void)out_size; (void)ws_size;
}
